// GaussianMixture_70884140253931
// MI455X (gfx1250) — matrix operand order repaired and checked by reading, not compiled, not run on hardware
//
#include <hip/hip_runtime.h>
#include <stddef.h>

// GMM log-likelihood, MI455X (gfx1250, wave32).
// ll_i = logsumexp_k( -0.5*||L_k^T x_i - L_k^T c_k||^2 + log coef_k ) - threshold
// Dominant GEMM done with v_wmma_f32_16x16x32_bf16 using split-bf16 (hi+lo)
// 3-pass emulation for ~f32 accuracy at bf16 matrix throughput.

typedef __attribute__((ext_vector_type(16))) __bf16        bf16x16;
typedef __attribute__((ext_vector_type(8)))  float         f32x8;
typedef __attribute__((ext_vector_type(8)))  unsigned int  u32x8;

#define GN 65536
#define GD 64
#define GK 32

// ---- bf16 helpers (round-to-nearest-even via bit trick) ----
__device__ __forceinline__ unsigned short bf16_rtne(float f) {
    unsigned int u = __float_as_uint(f);
    unsigned int r = u + 0x7FFFu + ((u >> 16) & 1u);
    return (unsigned short)(r >> 16);
}
__device__ __forceinline__ float bf16_to_f(unsigned short h) {
    return __uint_as_float(((unsigned int)h) << 16);
}

// ---- WMMA wrapper ----
__device__ __forceinline__ f32x8 wmma_bf16(u32x8 a, u32x8 b, f32x8 c) {
    return __builtin_amdgcn_wmma_f32_16x16x32_bf16(
        /*neg_a=*/false, __builtin_bit_cast(bf16x16, a),
        /*neg_b=*/false, __builtin_bit_cast(bf16x16, b),
        /*c_mod=*/(short)0, c, /*reuse_a=*/false, /*reuse_b=*/false);
}

// Build a 16x32 bf16 A fragment (hi and lo split) for this lane.
// Layout (CDNA5 ISA): lanes 0-15: M=lane, K pairs at kb=0; lanes 16-31: kb=8.
// VGPR j (0..7) holds K = base + 2j, 2j+1 where cols are
// [base .. base+7] (j=0..3) and [base+16 .. base+23] (j=4..7).
__device__ __forceinline__ void build_afrag(const float* __restrict__ xr, int base,
                                            u32x8& hi, u32x8& lo) {
    float v[16];
    const float4* p0 = (const float4*)(xr + base);
    const float4* p1 = (const float4*)(xr + base + 16);
    float4 a = p0[0], b = p0[1], c = p1[0], d = p1[1];
    v[0]=a.x; v[1]=a.y; v[2]=a.z; v[3]=a.w;
    v[4]=b.x; v[5]=b.y; v[6]=b.z; v[7]=b.w;
    v[8]=c.x; v[9]=c.y; v[10]=c.z; v[11]=c.w;
    v[12]=d.x; v[13]=d.y; v[14]=d.z; v[15]=d.w;
#pragma unroll
    for (int j = 0; j < 8; ++j) {
        float f0 = v[2*j], f1 = v[2*j+1];
        unsigned short h0 = bf16_rtne(f0), h1 = bf16_rtne(f1);
        unsigned short l0 = bf16_rtne(f0 - bf16_to_f(h0));
        unsigned short l1 = bf16_rtne(f1 - bf16_to_f(h1));
        hi[j] = (unsigned int)h0 | ((unsigned int)h1 << 16);
        lo[j] = (unsigned int)l0 | ((unsigned int)l1 << 16);
    }
}

// Load a 32x16 bf16 B fragment from LDS: lane holds column N = nt (passed as
// nt*16 + lane%16), VGPR j holds K = d + 2j, 2j+1 (j=0..3) and
// d + 16 + 2(j-4), +1 (j=4..7), with d = qbase + kb. Row stride 68 elems.
__device__ __forceinline__ u32x8 load_bfrag(const unsigned short* __restrict__ s,
                                            int d, int nt) {
    u32x8 r;
#pragma unroll
    for (int j = 0; j < 8; ++j) {
        int k0 = d + (j < 4 ? 2 * j : 16 + 2 * (j - 4));
        r[j] = (unsigned int)s[k0 * 68 + nt] | ((unsigned int)s[(k0 + 1) * 68 + nt] << 16);
    }
    return r;
}

// ---------------- Precompute: t_k = L_k^T c_k, logcoef_k ----------------
__global__ void gmm_precompute(const float* __restrict__ center,
                               const float* __restrict__ Lmat,
                               const float* __restrict__ weight,
                               float* __restrict__ t_out,
                               float* __restrict__ lc_out) {
    __shared__ float sL[64][65];
    __shared__ float sC[64];
    const int k = blockIdx.x;
    const int j = threadIdx.x;  // 0..63
    for (int i = 0; i < 64; ++i) sL[i][j] = Lmat[(size_t)k * 4096 + i * 64 + j];
    sC[j] = center[k * 64 + j];
    __syncthreads();
    float tj = 0.f;
    for (int d = 0; d < 64; ++d) tj = fmaf(sL[d][j], sC[d], tj);
    t_out[k * 64 + j] = tj;
    // Gaussian elimination, no pivoting (L = I + 0.05*noise: well-conditioned).
    for (int p = 0; p < 63; ++p) {
        __syncthreads();
        if (j > p) {
            float f = sL[j][p] / sL[p][p];
            for (int c = p; c < 64; ++c) sL[j][c] -= f * sL[p][c];
        }
    }
    __syncthreads();
    if (j == 0) {
        float logdet = 0.f;
        for (int i = 0; i < 64; ++i) logdet += logf(fabsf(sL[i][i]));
        float sumw = 0.f;
        for (int i = 0; i < GK; ++i) sumw += fabsf(weight[i]);
        // coef = pr * sqrt(det(L L^T)) = pr * |det L|
        lc_out[k] = logf(fabsf(weight[k])) - logf(sumw) + logdet;
    }
}

// ---------------- Main: WMMA GEMM + online logsumexp ----------------
__global__ void __launch_bounds__(256, 1)
gmm_main(const float* __restrict__ X,
         const float* __restrict__ Lmat,
         const float* __restrict__ t_ws,
         const float* __restrict__ lc_ws,
         const float* __restrict__ thr_p,
         float* __restrict__ out) {
    __shared__ __attribute__((aligned(16))) unsigned short sBhi[64 * 68];
    __shared__ __attribute__((aligned(16))) unsigned short sBlo[64 * 68];
    __shared__ float sT[64];
    __shared__ float sLC;

    const int tid  = threadIdx.x;
    const int lane = tid & 31;
    const int wave = tid >> 5;
    const int m0   = blockIdx.x * 256 + wave * 32;  // 32 rows per wave
    const int kb   = (lane & 16) ? 8 : 0;
    const int mrow = lane & 15;

    // A fragments for two 16-row tiles, hi/lo split, both K-chunks. Kept in VGPRs.
    u32x8 Ahi[2][2], Alo[2][2];
#pragma unroll
    for (int T = 0; T < 2; ++T) {
        const float* xr = X + (size_t)(m0 + T * 16 + mrow) * GD;
#pragma unroll
        for (int q = 0; q < 2; ++q)
            build_afrag(xr, q * 32 + kb, Ahi[T][q], Alo[T][q]);
    }

    float mrun[2][8], srun[2][8];
#pragma unroll
    for (int T = 0; T < 2; ++T)
#pragma unroll
        for (int r = 0; r < 8; ++r) { mrun[T][r] = -3.4e38f; srun[T][r] = 0.f; }

    for (int k = 0; k < GK; ++k) {
        __syncthreads();  // previous-k LDS consumers done
        // Stage L_k (64x64 f32) into LDS as bf16 hi/lo, row stride 68.
        for (int idx = tid; idx < 4096; idx += 256) {
            float f = Lmat[(size_t)k * 4096 + idx];
            unsigned short h = bf16_rtne(f);
            unsigned short l = bf16_rtne(f - bf16_to_f(h));
            int rr = idx >> 6, cc = idx & 63;
            sBhi[rr * 68 + cc] = h;
            sBlo[rr * 68 + cc] = l;
        }
        if (tid < 64) sT[tid] = t_ws[k * 64 + tid];
        if (tid == 0) sLC = lc_ws[k];
        __syncthreads();
        const float lc = sLC;

        float part[2][8];
#pragma unroll
        for (int T = 0; T < 2; ++T)
#pragma unroll
            for (int r = 0; r < 8; ++r) part[T][r] = 0.f;

#pragma unroll
        for (int nt = 0; nt < 4; ++nt) {  // output column tiles (D=64 -> 4x16)
            f32x8 acc0 = {0.f,0.f,0.f,0.f,0.f,0.f,0.f,0.f};
            f32x8 acc1 = {0.f,0.f,0.f,0.f,0.f,0.f,0.f,0.f};
#pragma unroll
            for (int q = 0; q < 2; ++q) {  // K chunks of 32
                u32x8 bhi = load_bfrag(sBhi, q * 32 + kb, nt * 16 + mrow);
                u32x8 blo = load_bfrag(sBlo, q * 32 + kb, nt * 16 + mrow);
                // split-f32 emulation: hi*hi + hi*lo + lo*hi
                acc0 = wmma_bf16(Ahi[0][q], bhi, acc0);
                acc1 = wmma_bf16(Ahi[1][q], bhi, acc1);
                acc0 = wmma_bf16(Ahi[0][q], blo, acc0);
                acc1 = wmma_bf16(Ahi[1][q], blo, acc1);
                acc0 = wmma_bf16(Alo[0][q], bhi, acc0);
                acc1 = wmma_bf16(Alo[1][q], bhi, acc1);
            }
            const float tc = sT[nt * 16 + mrow];  // t_c for this lane's column
#pragma unroll
            for (int r = 0; r < 8; ++r) {
                float d0 = acc0[r] - tc; part[0][r] = fmaf(d0, d0, part[0][r]);
                float d1 = acc1[r] - tc; part[1][r] = fmaf(d1, d1, part[1][r]);
            }
        }
        // Reduce 16 columns across each 16-lane half, then online logsumexp.
#pragma unroll
        for (int T = 0; T < 2; ++T)
#pragma unroll
            for (int r = 0; r < 8; ++r) {
                float v = part[T][r];
                v += __shfl_xor(v, 1, 32);
                v += __shfl_xor(v, 2, 32);
                v += __shfl_xor(v, 4, 32);
                v += __shfl_xor(v, 8, 32);
                float a  = fmaf(-0.5f, v, lc);
                float nm = fmaxf(mrun[T][r], a);
                srun[T][r] = srun[T][r] * __expf(mrun[T][r] - nm) + __expf(a - nm);
                mrun[T][r] = nm;
            }
    }

    const float thr = thr_p[0];
    if (mrow == 0) {  // lanes 0 and 16 hold the reduced values
        const int rbase = m0 + ((lane >> 4) ? 8 : 0);
#pragma unroll
        for (int T = 0; T < 2; ++T)
#pragma unroll
            for (int r = 0; r < 8; ++r)
                out[rbase + T * 16 + r] = mrun[T][r] + __logf(srun[T][r]) - thr;
    }
}

extern "C" void kernel_launch(void* const* d_in, const int* in_sizes, int n_in,
                              void* d_out, int out_size, void* d_ws, size_t ws_size,
                              hipStream_t stream) {
    const float* X      = (const float*)d_in[0];  // [N,D]
    const float* center = (const float*)d_in[1];  // [K,D]
    const float* Lmat   = (const float*)d_in[2];  // [K,D,D]
    const float* weight = (const float*)d_in[3];  // [K]
    const float* thr    = (const float*)d_in[4];  // scalar
    float* out = (float*)d_out;

    float* ws_t  = (float*)d_ws;        // K*D floats: t_k = L_k^T c_k
    float* ws_lc = ws_t + GK * GD;      // K floats: log coef_k

    gmm_precompute<<<GK, 64, 0, stream>>>(center, Lmat, weight, ws_t, ws_lc);
    gmm_main<<<GN / 256, 256, 0, stream>>>(X, Lmat, ws_t, ws_lc, thr, out);
}
